// RelationAwareAttentionHead_12558484374210
// MI455X (gfx1250) — matrix operand order repaired and checked by reading, not compiled, not run on hardware
//
#include <hip/hip_runtime.h>

typedef __attribute__((ext_vector_type(16))) _Float16 v16h;
typedef __attribute__((ext_vector_type(8)))  _Float16 v8h;
typedef __attribute__((ext_vector_type(8)))  float    v8f;
typedef __attribute__((ext_vector_type(4)))  unsigned int u32x4;
typedef __attribute__((ext_vector_type(8)))  int      i32x8;
typedef __attribute__((ext_vector_type(4)))  int      i32x4;

constexpr int B = 8, S = 1024, H = 768, D = 64;

static __device__ __forceinline__ v8f wmma_f16(v16h a, v16h b, v8f c) {
  // D = A(16x32 f16) * B(32x16 f16) + C(16x16 f32)
  return __builtin_amdgcn_wmma_f32_16x16x32_f16(false, a, false, b, (short)0, c, false, false);
}

// ---- LDS byte offset of a __shared__ object (for TDM descriptors) ----
typedef __attribute__((address_space(3))) char lds_char_t;
static __device__ __forceinline__ unsigned lds_addr_of(void* p) {
  return (unsigned)(unsigned long long)(lds_char_t*)p;
}

// ---- TDM: async DMA of a 2D fp32 tile (tile_d0 x tile_d1, row stride stride_d0)
//      from global memory into LDS.  D# packed per CDNA5 ISA ch.8.
//      6-arg toolchain form: (g0, g1, g2, g3, g4, cpol). ----
static __device__ __forceinline__ void tdm_load_2d_f32(
    const float* gaddr, unsigned lds_byte_off,
    unsigned tensor_d0, unsigned tensor_d1,
    unsigned tile_d0, unsigned tile_d1,
    unsigned stride_d0) {
  unsigned long long ga = (unsigned long long)gaddr;
  u32x4 g0;
  g0[0] = 1u;                                   // count=1 (valid), user mode, no gather
  g0[1] = lds_byte_off;                         // lds_addr (bytes)
  g0[2] = (unsigned)(ga & 0xFFFFFFFFu);         // global_addr[31:0]
  g0[3] = (unsigned)((ga >> 32) & 0x1FFFFFFu)   // global_addr[56:32]
        | (2u << 30);                           // type = 2 ("image")
  i32x8 g1;
  g1[0] = (int)(2u << 16);                      // workgroup_mask=0, data_size=2 (4 bytes)
  g1[1] = (int)((tensor_d0 & 0xFFFFu) << 16);   // tensor_dim0[15:0] @ bits 63:48
  g1[2] = (int)(((tensor_d0 >> 16) & 0xFFFFu) | ((tensor_d1 & 0xFFFFu) << 16));
  g1[3] = (int)(((tensor_d1 >> 16) & 0xFFFFu) | ((tile_d0 & 0xFFFFu) << 16));
  g1[4] = (int)(tile_d1 & 0xFFFFu);             // tile_dim1; tile_dim2=0 (2D)
  g1[5] = (int)stride_d0;                       // tensor_dim0_stride[31:0]
  g1[6] = 0;                                    // stride[47:32]; dim1_stride unused (2D)
  g1[7] = 0;
  i32x4 z4 = {0, 0, 0, 0};                      // groups 2/3 unused (<=2D tensor)
  i32x8 z8 = {0, 0, 0, 0, 0, 0, 0, 0};
  __builtin_amdgcn_tensor_load_to_lds(g0, g1, z4, z4, z8, 0);
}

// ---- fragment loaders (wave32; element i of lane half h holds k = 16h + i, same in A and B) ----
// A 16x32 f16: lanes 0-15 -> row=lane, K = 0..15; lanes 16-31 -> row=lane-16, K = 16..31
static __device__ __forceinline__ v16h load_a_f16(const _Float16* base, int ld, int k0, int lane) {
  const int r  = lane & 15;
  const int kh = (lane < 16) ? 0 : 16;
  const _Float16* p = base + (size_t)r * ld + k0 + kh;
  v8h lo = *(const v8h*)(p);
  v8h hi = *(const v8h*)(p + 8);
  v16h a;
#pragma unroll
  for (int i = 0; i < 8; ++i) { a[i] = lo[i]; a[8 + i] = hi[i]; }
  return a;
}

static __device__ __forceinline__ v16h load_a_f32cvt(const float* base, int ld, int k0, int lane) {
  const int r  = lane & 15;
  const int kh = (lane < 16) ? 0 : 16;
  const float* p = base + (size_t)r * ld + k0 + kh;
  v16h a;
#pragma unroll
  for (int i = 0; i < 8; ++i) { a[i] = (_Float16)p[i]; a[8 + i] = (_Float16)p[8 + i]; }
  return a;
}

// A with rows 8..15 zero-padded (M = batch dim, 8 valid rows), f16 source, row stride rstride
static __device__ __forceinline__ v16h load_a_pad8(const _Float16* base, size_t rstride, int k0, int lane) {
  const int r  = lane & 15;
  const int kh = (lane < 16) ? 0 : 16;
  v16h a;
  if (r < 8) {
    const _Float16* p = base + (size_t)r * rstride + k0 + kh;
#pragma unroll
    for (int i = 0; i < 8; ++i) { a[i] = p[i]; a[8 + i] = p[8 + i]; }
  } else {
#pragma unroll
    for (int i = 0; i < 16; ++i) a[i] = (_Float16)0.0f;
  }
  return a;
}

// B 32x16: lane = column; K rows have stride ld in memory
static __device__ __forceinline__ v16h load_b_f16(const _Float16* base, int ld, int k0, int lane) {
  const int c  = lane & 15;
  const int kb = k0 + ((lane < 16) ? 0 : 16);
  v16h b;
#pragma unroll
  for (int i = 0; i < 16; ++i) b[i] = base[(size_t)(kb + i) * ld + c];
  return b;
}

static __device__ __forceinline__ v16h load_b_f32cvt(const float* base, int ld, int k0, int lane) {
  const int c  = lane & 15;
  const int kb = k0 + ((lane < 16) ? 0 : 16);
  v16h b;
#pragma unroll
  for (int i = 0; i < 16; ++i) b[i] = (_Float16)base[(size_t)(kb + i) * ld + c];
  return b;
}

// B with K contiguous in memory, columns strided by ld (e.g. k^T)
static __device__ __forceinline__ v16h load_bT_f16(const _Float16* base, int ld, int k0, int lane) {
  const int c  = lane & 15;
  const int kb = k0 + ((lane < 16) ? 0 : 16);
  const _Float16* p = base + (size_t)c * ld + kb;
  v8h lo = *(const v8h*)(p);
  v8h hi = *(const v8h*)(p + 8);
  v16h b;
#pragma unroll
  for (int i = 0; i < 8; ++i) { b[i] = lo[i]; b[8 + i] = hi[i]; }
  return b;
}

// LDS variants (fp32 staged tile -> f16 fragment)
static __device__ __forceinline__ v16h load_bT_lds(const float* base, int ld, int k0, int lane) {
  const int c  = lane & 15;
  const int kb = k0 + ((lane < 16) ? 0 : 16);
  const float* p = base + c * ld + kb;
  v16h b;
#pragma unroll
  for (int i = 0; i < 16; ++i) b[i] = (_Float16)p[i];
  return b;
}

static __device__ __forceinline__ v16h load_b_lds(const float* base, int ld, int k0, int lane) {
  const int c  = lane & 15;
  const int kb = k0 + ((lane < 16) ? 0 : 16);
  v16h b;
#pragma unroll
  for (int i = 0; i < 16; ++i) b[i] = (_Float16)base[(kb + i) * ld + c];
  return b;
}

// ---- kernel 1: q/k/v projections  [B*S,H] @ [H,D] + bias -> f16 ----
__global__ __launch_bounds__(32) void proj_kernel(
    const float* __restrict__ x_q, const float* __restrict__ x_k, const float* __restrict__ x_v,
    const float* __restrict__ Wq, const float* __restrict__ bq,
    const float* __restrict__ Wk, const float* __restrict__ bk,
    const float* __restrict__ Wv, const float* __restrict__ bv,
    _Float16* __restrict__ qf, _Float16* __restrict__ kf, _Float16* __restrict__ vf) {
  const int lane = threadIdx.x;
  const int rt   = blockIdx.x;        // 16-row tile over B*S
  const int col0 = blockIdx.y * 16;   // D tile
  const int w    = blockIdx.z;
  const float* X    = (w == 0) ? x_q : (w == 1) ? x_k : x_v;
  const float* Wm   = (w == 0) ? Wq  : (w == 1) ? Wk  : Wv;
  const float* bias = (w == 0) ? bq  : (w == 1) ? bk  : bv;
  _Float16*    out  = (w == 0) ? qf  : (w == 1) ? kf  : vf;

  const float bval = bias[col0 + (lane & 15)];
  v8f c;
#pragma unroll
  for (int i = 0; i < 8; ++i) c[i] = bval;

  const float* xbase = X + (size_t)rt * 16 * H;
#pragma unroll 4
  for (int k0 = 0; k0 < H; k0 += 32) {
    v16h a = load_a_f32cvt(xbase, H, k0, lane);
    v16h b = load_b_f32cvt(Wm + col0, D, k0, lane);
    c = wmma_f16(a, b, c);
  }
  const int col   = col0 + (lane & 15);
  const int rbase = rt * 16 + ((lane < 16) ? 0 : 8);
#pragma unroll
  for (int i = 0; i < 8; ++i)
    out[(size_t)(rbase + i) * D + col] = (_Float16)c[i];
}

// ---- kernel 2: content scores q . k^T -> fp32 scores[B,S,S] ----
__global__ __launch_bounds__(32) void attn1_kernel(const _Float16* __restrict__ qf,
                                                   const _Float16* __restrict__ kf,
                                                   float* __restrict__ scores) {
  const int lane = threadIdx.x;
  const int b  = blockIdx.x >> 6;
  const int qt = blockIdx.x & 63;
  const int kt = blockIdx.y;
  const _Float16* abase = qf + (size_t)(b * S + qt * 16) * D;
  const _Float16* bbase = kf + (size_t)(b * S + kt * 16) * D;  // B[d][key]
  v8f c = {};
#pragma unroll
  for (int k0 = 0; k0 < D; k0 += 32) {
    v16h a  = load_a_f16(abase, D, k0, lane);
    v16h bm = load_bT_f16(bbase, D, k0, lane);
    c = wmma_f16(a, bm, c);
  }
  const int col = kt * 16 + (lane & 15);
  const int qr  = qt * 16 + ((lane < 16) ? 0 : 8);
#pragma unroll
  for (int i = 0; i < 8; ++i)
    scores[((size_t)b * S + (qr + i)) * S + col] = c[i];
}

// ---- kernel 3: relative-key bias scores; TDM-staged k_bias[q], GEMM over batch dim ----
__global__ __launch_bounds__(32) void attn2_kernel(const _Float16* __restrict__ qf,
                                                   const float* __restrict__ k_bias,
                                                   float* __restrict__ scores) {
  __shared__ float tile[2][64 * D];          // 2 x 16KB double buffer
  const int lane = threadIdx.x;
  const int qq   = blockIdx.x;
  const _Float16* abase = qf + (size_t)qq * D;      // batch stride S*D
  // A fragments (q vectors across batch) are reused for the whole row of keys
  v16h a0 = load_a_pad8(abase, (size_t)S * D, 0, lane);
  v16h a1 = load_a_pad8(abase, (size_t)S * D, 32, lane);
  const float* gbase = k_bias + (size_t)qq * S * D; // [key][d], d contiguous
  constexpr int NC = S / 64;                        // 16 chunks of 64 keys

  tdm_load_2d_f32(gbase, lds_addr_of(&tile[0][0]), D, 64, D, 64, D);
  for (int ch = 0; ch < NC; ++ch) {
    if (ch + 1 < NC) {
      tdm_load_2d_f32(gbase + (size_t)(ch + 1) * 64 * D,
                      lds_addr_of(&tile[(ch + 1) & 1][0]), D, 64, D, 64, D);
      __builtin_amdgcn_s_wait_tensorcnt(1);   // current chunk landed
    } else {
      __builtin_amdgcn_s_wait_tensorcnt(0);
    }
    const float* lt = &tile[ch & 1][0];
#pragma unroll
    for (int nt = 0; nt < 4; ++nt) {          // 4 key-tiles of 16 per chunk
      v8f c = {};
      v16h b0 = load_bT_lds(lt + nt * 16 * D, D, 0, lane);
      c = wmma_f16(a0, b0, c);
      v16h b1 = load_bT_lds(lt + nt * 16 * D, D, 32, lane);
      c = wmma_f16(a1, b1, c);
      if (lane < 16) {                        // D-tile rows 0..7 = batches
#pragma unroll
        for (int i = 0; i < 8; ++i)
          scores[((size_t)i * S + qq) * S + ch * 64 + nt * 16 + lane] += c[i];
      }
    }
  }
}

// ---- kernel 4: scale + mask + softmax, emit f16 weights ----
__global__ __launch_bounds__(256) void softmax_kernel(const float* __restrict__ scores,
                                                      const int* __restrict__ mask,
                                                      _Float16* __restrict__ wts) {
  __shared__ float red[256];
  const int row = blockIdx.x;     // b*S + q
  const int b   = row >> 10;
  const int t   = threadIdx.x;
  const float scale = 0.125f;     // 1/sqrt(64)
  float v[4];
  float m = -3.0e38f;
#pragma unroll
  for (int j = 0; j < 4; ++j) {
    const int k = t + j * 256;
    float s = scores[(size_t)row * S + k] * scale;
    s = (mask[b * S + k] == 0) ? -1.0e9f : s;
    v[j] = s;
    m = fmaxf(m, s);
  }
  red[t] = m; __syncthreads();
  for (int off = 128; off > 0; off >>= 1) {
    if (t < off) red[t] = fmaxf(red[t], red[t + off]);
    __syncthreads();
  }
  m = red[0]; __syncthreads();
  float sum = 0.f;
#pragma unroll
  for (int j = 0; j < 4; ++j) { v[j] = __expf(v[j] - m); sum += v[j]; }
  red[t] = sum; __syncthreads();
  for (int off = 128; off > 0; off >>= 1) {
    if (t < off) red[t] += red[t + off];
    __syncthreads();
  }
  const float inv = 1.0f / red[0];
#pragma unroll
  for (int j = 0; j < 4; ++j)
    wts[(size_t)row * S + t + j * 256] = (_Float16)(v[j] * inv);
}

// ---- kernel 5: v1 = w @ v  -> fp32 out[B,S,D] ----
__global__ __launch_bounds__(32) void v1_kernel(const _Float16* __restrict__ wts,
                                                const _Float16* __restrict__ vf,
                                                float* __restrict__ out) {
  const int lane = threadIdx.x;
  const int b  = blockIdx.x >> 6;
  const int qt = blockIdx.x & 63;
  const int d0 = blockIdx.y * 16;
  const _Float16* abase = wts + (size_t)(b * S + qt * 16) * S;
  const _Float16* bbase = vf + (size_t)b * S * D + d0;   // B[k][d]
  v8f c = {};
  for (int k0 = 0; k0 < S; k0 += 32) {
    v16h a  = load_a_f16(abase, S, k0, lane);
    v16h bm = load_b_f16(bbase, D, k0, lane);
    c = wmma_f16(a, bm, c);
  }
  const int col   = d0 + (lane & 15);
  const int rbase = (b * S + qt * 16) + ((lane < 16) ? 0 : 8);
#pragma unroll
  for (int i = 0; i < 8; ++i)
    out[(size_t)(rbase + i) * D + col] = c[i];
}

// ---- kernel 6: v2 = w . v_bias; TDM-staged v_bias[q], GEMM over batch dim, added into out ----
__global__ __launch_bounds__(32) void v2_kernel(const _Float16* __restrict__ wts,
                                                const float* __restrict__ v_bias,
                                                float* __restrict__ out) {
  __shared__ float tile[2][64 * D];          // 2 x 16KB double buffer
  const int lane = threadIdx.x;
  const int qq   = blockIdx.x;
  const _Float16* abase = wts + (size_t)qq * S;     // batch stride S*S
  const float* gbase = v_bias + (size_t)qq * S * D; // [key][d], d contiguous
  constexpr int NC = S / 64;
  v8f acc[4] = {};

  tdm_load_2d_f32(gbase, lds_addr_of(&tile[0][0]), D, 64, D, 64, D);
  for (int ch = 0; ch < NC; ++ch) {
    if (ch + 1 < NC) {
      tdm_load_2d_f32(gbase + (size_t)(ch + 1) * 64 * D,
                      lds_addr_of(&tile[(ch + 1) & 1][0]), D, 64, D, 64, D);
      __builtin_amdgcn_s_wait_tensorcnt(1);
    } else {
      __builtin_amdgcn_s_wait_tensorcnt(0);
    }
    const float* lt = &tile[ch & 1][0];
#pragma unroll
    for (int k0 = 0; k0 < 64; k0 += 32) {
      v16h a = load_a_pad8(abase, (size_t)S * S, ch * 64 + k0, lane);
#pragma unroll
      for (int dt = 0; dt < 4; ++dt) {
        v16h bm = load_b_lds(lt + dt * 16, D, k0, lane);
        acc[dt] = wmma_f16(a, bm, acc[dt]);
      }
    }
  }
  if (lane < 16) {
#pragma unroll
    for (int dt = 0; dt < 4; ++dt)
#pragma unroll
      for (int i = 0; i < 8; ++i)
        out[((size_t)i * S + qq) * D + dt * 16 + lane] += acc[dt][i];
  }
}

extern "C" void kernel_launch(void* const* d_in, const int* in_sizes, int n_in,
                              void* d_out, int out_size, void* d_ws, size_t ws_size,
                              hipStream_t stream) {
  const float* query  = (const float*)d_in[0];
  const float* key_   = (const float*)d_in[1];
  const float* value  = (const float*)d_in[2];
  const int*   mask   = (const int*)d_in[3];
  const float* Wq     = (const float*)d_in[4];
  const float* bq     = (const float*)d_in[5];
  const float* Wk     = (const float*)d_in[6];
  const float* bk     = (const float*)d_in[7];
  const float* Wv     = (const float*)d_in[8];
  const float* bv     = (const float*)d_in[9];
  const float* k_bias = (const float*)d_in[10];
  const float* v_bias = (const float*)d_in[11];
  float* out = (float*)d_out;

  // workspace: qf16|kf16|vf16 (1MB each), scores f32 (32MB), weights f16 (16MB)
  _Float16* qf = (_Float16*)d_ws;
  _Float16* kf = qf + (size_t)B * S * D;
  _Float16* vf = kf + (size_t)B * S * D;
  float* scores = (float*)(vf + (size_t)B * S * D);
  _Float16* wts = (_Float16*)(scores + (size_t)B * S * S);

  proj_kernel<<<dim3(B * S / 16, D / 16, 3), 32, 0, stream>>>(
      query, key_, value, Wq, bq, Wk, bk, Wv, bv, qf, kf, vf);
  attn1_kernel<<<dim3(B * S / 16, S / 16), 32, 0, stream>>>(qf, kf, scores);
  attn2_kernel<<<dim3(S), 32, 0, stream>>>(qf, k_bias, scores);
  softmax_kernel<<<dim3(B * S), 256, 0, stream>>>(scores, mask, wts);
  v1_kernel<<<dim3(B * S / 16, D / 16), 32, 0, stream>>>(wts, vf, out);
  v2_kernel<<<dim3(S), 32, 0, stream>>>(wts, v_bias, out);
}
